// MultiHeadAttention_70153995813322
// MI455X (gfx1250) — matrix operand order repaired and checked by reading, not compiled, not run on hardware
//
#include <hip/hip_runtime.h>

// MultiHeadAttention (cross-attn, causal) for gfx1250.
// B=4, S=2048, D=1024, H=16, HD=64.
// Pipeline: [weights fp32 -> f16^T] x4 ; QKV GEMMs (f16 WMMA, f32 acc) ;
// flash-attention (f16 WMMA, base-2 online softmax) ; output GEMM -> fp32.
// Round 2: 2x row-tile register blocking per wave (B-fragment reuse across
// two WMMAs), exp2-domain softmax, rcp normalize.

#define B_  4
#define S_  2048
#define D_  1024
#define H_  16
#define HD_ 64
#define M_  (B_ * S_)   // 8192 rows for the projection GEMMs

typedef __attribute__((ext_vector_type(16))) _Float16 v16h;
typedef __attribute__((ext_vector_type(8)))  _Float16 v8h;
typedef __attribute__((ext_vector_type(8)))  float    v8f;
typedef __attribute__((ext_vector_type(4)))  float    f32x4;

#define NEG_INF_F (-__builtin_inff())
// 0.125 (1/sqrt(HD)) folded with log2(e): scores scaled directly into the
// base-2 domain so softmax uses raw v_exp_f32 (exp2) with no extra mul.
#define SCALE_LOG2E (0.125f * 1.44269504088896340736f)

static __device__ __forceinline__ v8f wmma_f16(v16h a, v16h b, v8f c) {
  // D(16x16,f32) = A(16x32,f16) * B(32x16,f16) + C
  return __builtin_amdgcn_wmma_f32_16x16x32_f16(
      /*neg_a=*/false, a, /*neg_b=*/false, b,
      /*c_mod=*/(short)0, c, /*reuse_a=*/false, /*reuse_b=*/false);
}

// ---------------------------------------------------------------------------
// Kernel 1: W (fp32, [K=D][N=D]) -> W^T (f16, [N][K]) via LDS-tiled transpose.
// ---------------------------------------------------------------------------
__global__ __launch_bounds__(256)
void transpose_convert_kernel(const float* __restrict__ W, _Float16* __restrict__ WT) {
  __shared__ float tile[32][33];
  const int tx = threadIdx.x;          // 0..31
  const int ty = threadIdx.y;          // 0..7
  const int x0 = blockIdx.x * 32;      // n
  const int y0 = blockIdx.y * 32;      // k
#pragma unroll
  for (int i = 0; i < 32; i += 8)
    tile[ty + i][tx] = W[(size_t)(y0 + ty + i) * D_ + (x0 + tx)];
  __syncthreads();
#pragma unroll
  for (int i = 0; i < 32; i += 8)
    WT[(size_t)(x0 + ty + i) * D_ + (y0 + tx)] = (_Float16)tile[tx][ty + i];
}

// ---------------------------------------------------------------------------
// Kernel 2: GEMM  Y[M,N] = X[M,K] * (W^T)^T + bias.
//   IN_HALF : X is f16 (attn buffer) vs fp32 (data/context, converted in-reg)
//   OUT_QKV : store f16 in [B,H,S,HD] head layout vs fp32 flat [M,N]
// Block: 128 threads (4 waves), tile 128x64, K-chunks of 32.
// Each wave owns TWO 16-row tiles -> every B fragment feeds 2 WMMAs.
// ---------------------------------------------------------------------------
template <bool IN_HALF, bool OUT_QKV>
__global__ __launch_bounds__(128)
void gemm_kernel(const void* __restrict__ Xv, const _Float16* __restrict__ WT,
                 const float* __restrict__ bias, void* __restrict__ Yv) {
  __shared__ _Float16 Wt_s[64][32];    // [n_local][k_local]

  const int tid  = threadIdx.x;
  const int lane = tid & 31;
  const int wave = tid >> 5;
  const int lg   = lane & 15;          // col (B/D) or row (A) index within tile
  const int lh   = lane >> 4;          // half-wave select
  const int mbase = blockIdx.y * 128;
  const int nbase = blockIdx.x * 64;

  const float*    Xf = (const float*)Xv;
  const _Float16* Xh = (const _Float16*)Xv;

  v8f acc[2][4];
#pragma unroll
  for (int rt = 0; rt < 2; ++rt)
#pragma unroll
    for (int t = 0; t < 4; ++t) acc[rt][t] = {};

  const int snl  = tid & 63;           // staging: row of W^T tile
  const int sseg = tid >> 6;           // staging: 16-half segment

  for (int kc = 0; kc < D_; kc += 32) {
    __syncthreads();
    {  // stage W^T tile (f16, contiguous) into LDS
      const v8h* src = (const v8h*)(WT + (size_t)(nbase + snl) * D_ + kc + sseg * 16);
      v8h* dst = (v8h*)(&Wt_s[snl][sseg * 16]);
      dst[0] = src[0];
      dst[1] = src[1];
    }
    __syncthreads();

    // A fragments straight from global (ISA 16-bit A layout: lane=row,
    // elems 0..7 -> K = kh..kh+7, elems 8..15 -> K = kh+16..kh+23, kh = lh*8)
    v16h a[2];
    const int k0 = kc + lh * 8;
#pragma unroll
    for (int rt = 0; rt < 2; ++rt) {
      const int row = mbase + wave * 32 + rt * 16 + lg;
      if constexpr (IN_HALF) {
        const _Float16* xp = Xh + (size_t)row * D_ + k0;
        ((v8h*)&a[rt])[0] = *(const v8h*)(xp);
        ((v8h*)&a[rt])[1] = *(const v8h*)(xp + 16);
      } else {
        const float* xp = Xf + (size_t)row * D_ + k0;
        f32x4 x0 = ((const f32x4*)xp)[0];
        f32x4 x1 = ((const f32x4*)xp)[1];
        f32x4 x2 = ((const f32x4*)(xp + 16))[0];
        f32x4 x3 = ((const f32x4*)(xp + 16))[1];
#pragma unroll
        for (int i = 0; i < 4; ++i) {
          a[rt][i]      = (_Float16)x0[i];
          a[rt][4 + i]  = (_Float16)x1[i];
          a[rt][8 + i]  = (_Float16)x2[i];
          a[rt][12 + i] = (_Float16)x3[i];
        }
        if (kc + 32 < D_) __builtin_prefetch(xp + 32, 0, 1);  // global_prefetch_b8
      }
    }

#pragma unroll
    for (int t = 0; t < 4; ++t) {
      // B fragment: lane col n = t*16+lg, elems 0..7 -> k = lh*8+i, 8..15 -> k = lh*8+16+i
      v16h b;
      const _Float16* wp = &Wt_s[t * 16 + lg][lh * 8];
      ((v8h*)&b)[0] = *(const v8h*)(wp);
      ((v8h*)&b)[1] = *(const v8h*)(wp + 16);
#pragma unroll
      for (int rt = 0; rt < 2; ++rt)        // reuse B across both row tiles
        acc[rt][t] = wmma_f16(a[rt], b, acc[rt][t]);
    }
  }

  // Epilogue: D layout -> lane holds col n = lg, rows m = r + 8*lh.
#pragma unroll
  for (int rt = 0; rt < 2; ++rt) {
#pragma unroll
    for (int t = 0; t < 4; ++t) {
      const int n  = nbase + t * 16 + lg;
      const float bn = bias[n];
#pragma unroll
      for (int r = 0; r < 8; ++r) {
        const int m = mbase + wave * 32 + rt * 16 + r + 8 * lh;
        const float v = acc[rt][t][r] + bn;
        if constexpr (OUT_QKV) {
          const int b  = m >> 11;        // m / S_
          const int s  = m & (S_ - 1);
          const int h  = n >> 6;         // n / HD_
          const int hd = n & 63;
          ((_Float16*)Yv)[(((size_t)(b * H_ + h)) * S_ + s) * HD_ + hd] = (_Float16)v;
        } else {
          ((float*)Yv)[(size_t)m * D_ + n] = v;
        }
      }
    }
  }
}

// ---------------------------------------------------------------------------
// Kernel 3: causal flash attention per (b*H+h, 128-row q block).
// 128 threads = 4 waves; each wave owns TWO 16-row Q stripes (32 rows),
// so every K/V B-fragment feeds two WMMAs. Online softmax in base-2 domain.
// Output written f16 into attn buffer laid out [B,S,D] for the final GEMM.
// ---------------------------------------------------------------------------
__global__ __launch_bounds__(128)
void attention_kernel(const _Float16* __restrict__ Qg, const _Float16* __restrict__ Kg,
                      const _Float16* __restrict__ Vg, _Float16* __restrict__ Og) {
  __shared__ _Float16 Kt[32][64];      // [kv][d]   (B-frag for QK^T contiguous)
  __shared__ _Float16 Vt[64][32];      // [d][kv]   (B-frag for PV contiguous)
  __shared__ _Float16 Pb[4][32][32];   // per-wave P tiles (D-layout -> A-layout)

  const int tid  = threadIdx.x;
  const int lane = tid & 31;
  const int wave = tid >> 5;
  const int lg   = lane & 15;
  const int lh   = lane >> 4;

  const int bh    = blockIdx.y;        // b*H + h
  const int qbase = blockIdx.x * 128;
  const int q0    = qbase + wave * 32; // wave's 32-row stripe
  const size_t base = (size_t)bh * S_ * HD_;

  // Q A-fragments: two row tiles x two 32-wide d chunks, loaded once.
  v16h qa[2][2];
#pragma unroll
  for (int rt = 0; rt < 2; ++rt) {
    const _Float16* qrow = Qg + base + (size_t)(q0 + rt * 16 + lg) * HD_;
#pragma unroll
    for (int c = 0; c < 2; ++c) {
      ((v8h*)&qa[rt][c])[0] = *(const v8h*)(qrow + c * 32 + lh * 8);
      ((v8h*)&qa[rt][c])[1] = *(const v8h*)(qrow + c * 32 + lh * 8 + 16);
    }
  }

  float mrow[2][8], lrow[2][8];
#pragma unroll
  for (int rt = 0; rt < 2; ++rt)
#pragma unroll
    for (int r = 0; r < 8; ++r) { mrow[rt][r] = NEG_INF_F; lrow[rt][r] = 0.f; }
  v8f acc[2][4];
#pragma unroll
  for (int rt = 0; rt < 2; ++rt)
#pragma unroll
    for (int t = 0; t < 4; ++t) acc[rt][t] = {};

  const int jmax = qbase / 32 + 3;     // last kv block touching the diagonal
  for (int j = 0; j <= jmax; ++j) {
    const int kv0 = j * 32;
    __syncthreads();
    {  // stage K tile [32][64]
      const int kr = tid >> 2, ko = (tid & 3) * 16;
      const v8h* src = (const v8h*)(Kg + base + (size_t)(kv0 + kr) * HD_ + ko);
      v8h* dst = (v8h*)(&Kt[kr][ko]);
      dst[0] = src[0]; dst[1] = src[1];
    }
    {  // stage V tile transposed -> Vt[d][kv]
      const int vr = tid >> 2, vo = (tid & 3) * 16;
      const v8h* src = (const v8h*)(Vg + base + (size_t)(kv0 + vr) * HD_ + vo);
      v8h v0 = src[0], v1 = src[1];
#pragma unroll
      for (int e = 0; e < 8; ++e) {
        Vt[vo + e][vr]     = v0[e];
        Vt[vo + 8 + e][vr] = v1[e];
      }
    }
    __syncthreads();

    // U = (Q K^T) * 0.125 * log2(e), causal-masked. Two col tiles x two
    // row tiles; each K B-fragment pair feeds both row tiles.
    float p[2][2][8];                  // [rt][t2][r], base-2 logits
#pragma unroll
    for (int t2 = 0; t2 < 2; ++t2) {
      v16h kb[2];
#pragma unroll
      for (int c = 0; c < 2; ++c) {    // B-frag: lane col = kv col, elems -> d
        const _Float16* kp = &Kt[t2 * 16 + lg][c * 32 + lh * 8];
        ((v8h*)&kb[c])[0] = *(const v8h*)(kp);
        ((v8h*)&kb[c])[1] = *(const v8h*)(kp + 16);
      }
      const int col = kv0 + t2 * 16 + lg;
#pragma unroll
      for (int rt = 0; rt < 2; ++rt) {
        v8f s = {};
        s = wmma_f16(qa[rt][0], kb[0], s);
        s = wmma_f16(qa[rt][1], kb[1], s);
#pragma unroll
        for (int r = 0; r < 8; ++r) {
          const int rowm = q0 + rt * 16 + r + 8 * lh;
          float u = s[r] * SCALE_LOG2E;
          if (col > rowm) u = NEG_INF_F;       // causal mask
          p[rt][t2][r] = u;
        }
      }
    }

#pragma unroll
    for (int rt = 0; rt < 2; ++rt) {
      float bm[8];
#pragma unroll
      for (int r = 0; r < 8; ++r) bm[r] = fmaxf(p[rt][0][r], p[rt][1][r]);
      // row max across the 16 column lanes (each half-wave owns its 8 rows)
#pragma unroll
      for (int off = 1; off < 16; off <<= 1)
#pragma unroll
        for (int r = 0; r < 8; ++r)
          bm[r] = fmaxf(bm[r], __shfl_xor(bm[r], off, 32));

      float alpha[8], rs[8];
#pragma unroll
      for (int r = 0; r < 8; ++r) {
        const float mn = fmaxf(mrow[rt][r], bm[r]);
        alpha[r] = __builtin_amdgcn_exp2f(mrow[rt][r] - mn);  // 2^(-inf)=0 first time
        mrow[rt][r] = mn;
        rs[r] = 0.f;
      }
      // P = 2^(U - m); row sums; spill P to per-wave LDS (D- -> A-layout)
#pragma unroll
      for (int t2 = 0; t2 < 2; ++t2)
#pragma unroll
        for (int r = 0; r < 8; ++r) {
          const float e = __builtin_amdgcn_exp2f(p[rt][t2][r] - mrow[rt][r]);
          rs[r] += e;
          Pb[wave][rt * 16 + r + 8 * lh][t2 * 16 + lg] = (_Float16)e;
        }
#pragma unroll
      for (int off = 1; off < 16; off <<= 1)
#pragma unroll
        for (int r = 0; r < 8; ++r)
          rs[r] += __shfl_xor(rs[r], off, 32);
#pragma unroll
      for (int r = 0; r < 8; ++r) lrow[rt][r] = lrow[rt][r] * alpha[r] + rs[r];
#pragma unroll
      for (int t = 0; t < 4; ++t)
#pragma unroll
        for (int r = 0; r < 8; ++r) acc[rt][t][r] *= alpha[r];
    }

    // PV: A-frags of P from LDS (same-wave DS ops are in-order);
    // each V B-fragment feeds both row tiles.
    v16h pa[2];
#pragma unroll
    for (int rt = 0; rt < 2; ++rt) {
      const _Float16* pp = &Pb[wave][rt * 16 + lg][lh * 8];
      ((v8h*)&pa[rt])[0] = *(const v8h*)(pp);
      ((v8h*)&pa[rt])[1] = *(const v8h*)(pp + 16);
    }
#pragma unroll
    for (int t = 0; t < 4; ++t) {
      v16h vb;  // B-frag: lane col = d = t*16+lg, elems 0..7 -> kv = lh*8+i, 8..15 -> kv = lh*8+16+i
      const _Float16* vp = &Vt[t * 16 + lg][lh * 8];
      ((v8h*)&vb)[0] = *(const v8h*)(vp);
      ((v8h*)&vb)[1] = *(const v8h*)(vp + 16);
#pragma unroll
      for (int rt = 0; rt < 2; ++rt)
        acc[rt][t] = wmma_f16(pa[rt], vb, acc[rt][t]);
    }
  }

  // Normalize (fast rcp) and store f16 into [B,S,D] attn buffer.
  const int b = bh >> 4;
  const int h = bh & 15;
#pragma unroll
  for (int rt = 0; rt < 2; ++rt) {
#pragma unroll
    for (int r = 0; r < 8; ++r) {
      const float inv = __builtin_amdgcn_rcpf(lrow[rt][r]);
      const int srow  = q0 + rt * 16 + r + 8 * lh;
      _Float16* op = Og + ((size_t)b * S_ + srow) * D_ + h * HD_;
#pragma unroll
      for (int t = 0; t < 4; ++t)
        op[t * 16 + lg] = (_Float16)(acc[rt][t][r] * inv);
    }
  }
}

// ---------------------------------------------------------------------------
// Host launcher
// ---------------------------------------------------------------------------
extern "C" void kernel_launch(void* const* d_in, const int* in_sizes, int n_in,
                              void* d_out, int out_size, void* d_ws, size_t ws_size,
                              hipStream_t stream) {
  const float* data    = (const float*)d_in[0];
  const float* context = (const float*)d_in[1];
  const float* Wq = (const float*)d_in[2];
  const float* bq = (const float*)d_in[3];
  const float* Wk = (const float*)d_in[4];
  const float* bk = (const float*)d_in[5];
  const float* Wv = (const float*)d_in[6];
  const float* bv = (const float*)d_in[7];
  const float* Wo = (const float*)d_in[8];
  const float* bo = (const float*)d_in[9];

  // Workspace layout (f16 elements). Total ~72 MB.
  constexpr size_t QKV_N = (size_t)B_ * H_ * S_ * HD_;  // 8388608
  constexpr size_t ATT_N = (size_t)B_ * S_ * D_;        // 8388608
  constexpr size_t WT_N  = (size_t)D_ * D_;             // 1048576
  _Float16* ws  = (_Float16*)d_ws;
  _Float16* Qb  = ws;
  _Float16* Kb  = Qb + QKV_N;
  _Float16* Vb  = Kb + QKV_N;
  _Float16* Ab  = Vb + QKV_N;
  _Float16* WqT = Ab + ATT_N;
  _Float16* WkT = WqT + WT_N;
  _Float16* WvT = WkT + WT_N;
  _Float16* WoT = WvT + WT_N;
  (void)ws_size; (void)in_sizes; (void)n_in; (void)out_size;

  // 1) weights -> f16 transposed
  dim3 tb(32, 8), tg(D_ / 32, D_ / 32);
  transpose_convert_kernel<<<tg, tb, 0, stream>>>(Wq, WqT);
  transpose_convert_kernel<<<tg, tb, 0, stream>>>(Wk, WkT);
  transpose_convert_kernel<<<tg, tb, 0, stream>>>(Wv, WvT);
  transpose_convert_kernel<<<tg, tb, 0, stream>>>(Wo, WoT);

  // 2) Q/K/V projections (fp32 in, f16 head-layout out)
  dim3 gb(128), gg(D_ / 64, M_ / 128);
  gemm_kernel<false, true><<<gg, gb, 0, stream>>>(data,    WqT, bq, Qb);
  gemm_kernel<false, true><<<gg, gb, 0, stream>>>(context, WkT, bk, Kb);
  gemm_kernel<false, true><<<gg, gb, 0, stream>>>(context, WvT, bv, Vb);

  // 3) causal flash attention (128-row q blocks)
  dim3 ab(128), ag(S_ / 128, B_ * H_);
  attention_kernel<<<ag, ab, 0, stream>>>(Qb, Kb, Vb, Ab);

  // 4) output projection (f16 in, fp32 out + bias)
  gemm_kernel<true, false><<<gg, gb, 0, stream>>>(Ab, WoT, bo, (float*)d_out);
}
